// CustomMoBAAttention_45492293599511
// MI455X (gfx1250) — matrix operand order repaired and checked by reading, not compiled, not run on hardware
//
#include <hip/hip_runtime.h>
#include <stdint.h>

// ---------------------------------------------------------------------------
// MoBA attention for MI455X (gfx1250), wave32 + WMMA bf16 + TDM/async-LDS.
// Pipeline: cvt/transpose -> QKV GEMM (WMMA) -> RoPE+relayout (K transposed)
//           -> block reps -> top-3 block select -> flash block-sparse
//           attention (WMMA, TDM-staged LDS tiles) -> output GEMM (WMMA).
// ---------------------------------------------------------------------------

#define HIDDEN 768
#define HEADS  12
#define HDIM   64
#define SEQ    2048
#define BATCH  2
#define BLK    256
#define NB     8
#define ROWS   (BATCH * SEQ)   // 4096

typedef __attribute__((ext_vector_type(16))) __bf16 v16bf;
typedef __attribute__((ext_vector_type(8)))  float  v8f;
typedef __attribute__((ext_vector_type(4)))  unsigned int u32x4;
typedef __attribute__((ext_vector_type(8)))  int    i32x8;
typedef __attribute__((ext_vector_type(4)))  int    i32x4;
typedef __attribute__((ext_vector_type(4)))  int    v4i;

typedef __attribute__((address_space(1))) v4i* g_v4i_p;   // global int4*
typedef __attribute__((address_space(3))) v4i* l_v4i_p;   // LDS int4*

// --- CDNA5 feature detection -----------------------------------------------
#if defined(__has_builtin)
#  if __has_builtin(__builtin_amdgcn_tensor_load_to_lds)
#    define HAVE_TDM 1
#  endif
#  if __has_builtin(__builtin_amdgcn_global_load_async_to_lds_b128)
#    define HAVE_ASYNC 1
#  endif
#endif

#if defined(__has_builtin)
#  if __has_builtin(__builtin_amdgcn_s_wait_asynccnt)
#    define WAIT_ASYNC0() __builtin_amdgcn_s_wait_asynccnt(0)
#  else
#    define WAIT_ASYNC0() asm volatile("s_wait_asynccnt 0x0" ::: "memory")
#  endif
#  if __has_builtin(__builtin_amdgcn_s_wait_tensorcnt)
#    define WAIT_TENSOR0() __builtin_amdgcn_s_wait_tensorcnt(0)
#  else
#    define WAIT_TENSOR0() asm volatile("s_wait_tensorcnt 0x0" ::: "memory")
#  endif
#else
#  define WAIT_ASYNC0()  asm volatile("s_wait_asynccnt 0x0" ::: "memory")
#  define WAIT_TENSOR0() asm volatile("s_wait_tensorcnt 0x0" ::: "memory")
#endif

union FragB { v16bf v; uint32_t d[8]; uint16_t h[16]; };

__device__ __forceinline__ uint16_t f2bf(float f) {
    union { float f; uint32_t u; } a; a.f = f;
    uint32_t r = a.u + 0x7fffu + ((a.u >> 16) & 1u);   // round-to-nearest-even
    return (uint16_t)(r >> 16);
}
__device__ __forceinline__ float bf2f(uint16_t h) {
    union { uint32_t u; float f; } a; a.u = ((uint32_t)h) << 16; return a.f;
}
__device__ __forceinline__ v8f wmma_bf16(const FragB& A, const FragB& B, v8f C) {
    return __builtin_amdgcn_wmma_f32_16x16x32_bf16(false, A.v, false, B.v,
                                                   (short)0, C, false, false);
}
// 16-bit A-matrix 16x32 K-offset for VGPR v in this lane (ISA 7.12.2).
__device__ __forceinline__ int a_koff(int v, int lane) {
    int kk = (v < 4) ? (2 * v) : (16 + 2 * (v - 4));
    return kk + ((lane >> 4) << 3);
}

#if defined(HAVE_ASYNC)
__device__ __forceinline__ void async_b128(const void* gsrc, void* ldst) {
    __builtin_amdgcn_global_load_async_to_lds_b128(
        (g_v4i_p)(uintptr_t)gsrc, (l_v4i_p)ldst, 0, 0);
}
#endif

#if defined(HAVE_TDM)
// Issue one TDM 2-D tile load: LDS[lds_off..] <= tile_h rows x tile_w elems
// (2-byte elements), global row stride row_stride elems.  D# layout per
// CDNA5 ISA; groups 2/3 zero (<=2-D tensor).
__device__ __forceinline__ void tdm_load_2d(unsigned lds_off, const void* gaddr,
                                            unsigned tile_w, unsigned tile_h,
                                            unsigned row_stride,
                                            unsigned tdim0, unsigned tdim1) {
    unsigned long long ga = (unsigned long long)(size_t)gaddr;
    u32x4 g0 = {};
    g0[0] = 1u;                                         // count=1, user mode
    g0[1] = lds_off;                                    // lds_addr @63:32
    g0[2] = (unsigned)ga;                               // global_addr lo @95:64
    g0[3] = (unsigned)((ga >> 32) & 0x1ffffffu) | (2u << 30);  // ga hi | type=2
    i32x8 g1 = {};
    g1[0] = (int)(1u << 16);                            // data_size=2B @17:16
    g1[1] = (int)((tdim0 & 0xffffu) << 16);             // tensor_dim0 lo @63:48
    g1[2] = (int)(((tdim0 >> 16) & 0xffffu) |           // tensor_dim0 hi @79:64
                  ((tdim1 & 0xffffu) << 16));           // tensor_dim1 lo @95:80
    g1[3] = (int)(((tdim1 >> 16) & 0xffffu) |           // tensor_dim1 hi @111:96
                  ((tile_w & 0xffffu) << 16));          // tile_dim0 @127:112
    g1[4] = (int)(tile_h & 0xffffu);                    // tile_dim1 @143:128
    g1[5] = (int)row_stride;                            // dim0_stride lo @191:160
    i32x4 g2 = {}, g3 = {};
    i32x8 g4 = {};
    __builtin_amdgcn_tensor_load_to_lds(g0, g1, g2, g3, g4, 0);
}
#endif

// ---------------------------------------------------------------------------
__global__ __launch_bounds__(256) void cvt_f32_bf16(const float* __restrict__ in,
                                                    uint16_t* __restrict__ out, int n) {
    int t = blockIdx.x * blockDim.x + threadIdx.x;
    if (t < n) out[t] = f2bf(in[t]);
}

// W[N,K] fp32 -> Wt[K,N] bf16 (GEMM B k-slab rows are contiguous in n)
__global__ __launch_bounds__(256) void transposeW(const float* __restrict__ W,
                                                  uint16_t* __restrict__ Wt) {
    int t = blockIdx.x * blockDim.x + threadIdx.x;      // over 768*768
    int k = t / HIDDEN, n = t % HIDDEN;
    Wt[t] = f2bf(W[(size_t)n * HIDDEN + k]);
}

// ---------------------------------------------------------------------------
// C[M,N] (f32) = A[M,K] (bf16) * Bt[K,N] (bf16).  128(M) x 64(N) per WG,
// 8 waves; Bt k-slab staged in LDS (async when available).
__global__ __launch_bounds__(256) void gemm_bf16(const uint16_t* __restrict__ A,
                                                 const uint16_t* __restrict__ Bt,
                                                 float* __restrict__ C,
                                                 int M, int N, int K) {
    __shared__ uint16_t Bs[32 * 64];
    const int lane = threadIdx.x & 31, wave = threadIdx.x >> 5;
    const int rowBase = blockIdx.y * 128 + wave * 16;
    const int colBase = blockIdx.x * 64;
    const int arow = rowBase + (lane & 15);
    v8f acc[4] = {};

    for (int k0 = 0; k0 < K; k0 += 32) {
        { // stage Bt[k0..k0+31][colBase..colBase+63] (4 KB), 1 x b128 / thread
            int t = threadIdx.x;
            int r = t >> 3, c8 = (t & 7) * 8;
            const uint16_t* src = &Bt[(size_t)(k0 + r) * N + colBase + c8];
#if defined(HAVE_ASYNC)
            async_b128(src, &Bs[r * 64 + c8]);
            WAIT_ASYNC0();
#else
            *(uint4*)&Bs[r * 64 + c8] = *(const uint4*)src;
#endif
        }
        __syncthreads();
        if (k0 + 32 < K)
            __builtin_prefetch(&A[(size_t)arow * K + k0 + 32], 0, 3);
        FragB af;
#pragma unroll
        for (int v = 0; v < 8; ++v)
            af.d[v] = *(const uint32_t*)&A[(size_t)arow * K + k0 + a_koff(v, lane)];
#pragma unroll
        for (int nt = 0; nt < 4; ++nt) {
            FragB bf;                            // B: n = lane%16, k per element (table order)
#pragma unroll
            for (int s = 0; s < 16; ++s)
                bf.h[s] = Bs[(a_koff(s >> 1, lane) + (s & 1)) * 64 + nt * 16 + (lane & 15)];
            acc[nt] = wmma_bf16(af, bf, acc[nt]);
        }
        __syncthreads();
    }
    const int hi8 = (lane >> 4) << 3;
#pragma unroll
    for (int nt = 0; nt < 4; ++nt)
#pragma unroll
        for (int r = 0; r < 8; ++r)
            C[(size_t)(rowBase + r + hi8) * N + colBase + nt * 16 + (lane & 15)] =
                acc[nt][r];
}

// ---------------------------------------------------------------------------
// RoPE (optional) + relayout.  transposed=0: [B*H, S, D]; transposed=1:
// [B*H, D, S] (used for K so attention/TDM get unit-stride key runs).
__global__ __launch_bounds__(256) void rope_layout(const float* __restrict__ X,
                                                   uint16_t* __restrict__ O,
                                                   int doRope, int transposed) {
    int t = blockIdx.x * blockDim.x + threadIdx.x;      // B*S*H*D
    int d = t & 63;
    int rest = t >> 6;
    int h = rest % HEADS;
    int rest2 = rest / HEADS;
    int s = rest2 & (SEQ - 1);
    int b = rest2 >> 11;
    size_t inb = ((size_t)(b * SEQ + s)) * HIDDEN + h * HDIM;
    float x = X[inb + d];
    float val = x;
    if (doRope) {
        int d2 = d & 31;
        float inv = __expf(-0.28782313662425572f * (float)d2); // 10000^(-2d2/64)
        float ang = (float)s * inv;
        float c = __cosf(ang), sn = __sinf(ang);
        float partner = X[inb + (d < 32 ? d + 32 : d - 32)];
        val = x * c + (d < 32 ? -partner : partner) * sn;
    }
    int bh = b * HEADS + h;
    size_t oi = transposed ? ((size_t)bh * HDIM + d) * SEQ + s
                           : ((size_t)bh * SEQ + s) * HDIM + d;
    O[oi] = f2bf(val);
}

// ---------------------------------------------------------------------------
// Block representatives from transposed K: mean over each 256-key block.
__global__ __launch_bounds__(256) void block_reps(const uint16_t* __restrict__ Kt,
                                                  float* __restrict__ reps) {
    int t = blockIdx.x * blockDim.x + threadIdx.x;      // 24*8*64
    int d = t & 63;
    int blk = (t >> 6) & 7;
    int bh = t >> 9;
    const uint16_t* p = Kt + ((size_t)bh * HDIM + d) * SEQ + blk * BLK;
    float s = 0.f;
    for (int i = 0; i < BLK; ++i) s += bf2f(p[i]);      // unit stride
    reps[(size_t)(bh * NB + blk) * HDIM + d] = s * (1.0f / 256.0f);
}

// ---------------------------------------------------------------------------
// Top-3 block selection per query (mirrors jax.lax.top_k tie-breaking,
// including the -inf-future-block selection quirk when cur < 2).
__global__ __launch_bounds__(256) void block_select(const uint16_t* __restrict__ Qh,
                                                    const float* __restrict__ reps,
                                                    uint8_t* __restrict__ mask) {
    int t = blockIdx.x * blockDim.x + threadIdx.x;      // 24*2048
    int s = t & (SEQ - 1);
    int bh = t >> 11;
    int cur = s >> 8;
    const uint16_t* q = Qh + ((size_t)bh * SEQ + s) * HDIM;
    float sc[NB];
#pragma unroll
    for (int blk = 0; blk < NB; ++blk) {
        if (blk > cur)      { sc[blk] = -3.0e38f; continue; }
        if (blk == cur)     { sc[blk] =  3.4e38f; continue; }
        const float* r = reps + (bh * NB + blk) * HDIM;
        float dot = 0.f;
        for (int d = 0; d < HDIM; ++d) dot += bf2f(q[d]) * r[d];
        sc[blk] = dot * 0.125f;
    }
    unsigned bits = 0;
#pragma unroll
    for (int pick = 0; pick < 3; ++pick) {
        int bi = 0; float bv = sc[0];
#pragma unroll
        for (int j = 1; j < NB; ++j)
            if (sc[j] > bv) { bv = sc[j]; bi = j; }
        bits |= 1u << bi;
        sc[bi] = -3.3e38f;   // below any real/future score; keeps tie order
    }
    mask[t] = (uint8_t)bits;
}

// ---------------------------------------------------------------------------
// Flash-style block-sparse attention.  WG = 128 threads = 4 waves, each wave
// owns a 16-query tile.  Per 128-key chunk, K^T ([64 d][128 key]) and V
// ([128 key][64 d]) tiles staged to LDS via TDM / async-LDS / sync copy.
__global__ __launch_bounds__(128) void moba_attn(const uint16_t* __restrict__ Qh,
                                                 const uint16_t* __restrict__ Kt,
                                                 const uint16_t* __restrict__ Vh,
                                                 const uint8_t* __restrict__ mask,
                                                 uint16_t* __restrict__ AO) {
    __shared__ uint16_t Ks[HDIM * 128];      // K^T tile: [d][key], 16 KB
    __shared__ uint16_t Vs[128 * HDIM];      // V tile:   [key][d], 16 KB
    __shared__ uint16_t Ps[4][16 * 32];      // per-wave P staging (C->A relayout)
    __shared__ unsigned selOr;

    const int lane = threadIdx.x & 31, wave = threadIdx.x >> 5;
    const int bh = blockIdx.y;
    const int b = bh / HEADS, h = bh % HEADS;
    const int qtile = blockIdx.x;                    // 32 tiles of 64 queries
    const int qbase = qtile * 64 + wave * 16;
    const int cur = (qtile * 64) >> 8;               // uniform over the WG
    const size_t qvbase = (size_t)bh * SEQ * HDIM;   // Qh / Vh head base
    const size_t ktbase = (size_t)bh * HDIM * SEQ;   // Kt head base
    const int hi8 = (lane >> 4) << 3;

    if (threadIdx.x == 0) selOr = 0;
    __syncthreads();
    if (threadIdx.x < 64)
        atomicOr(&selOr, (unsigned)mask[(size_t)bh * SEQ + qtile * 64 + threadIdx.x]);
    __syncthreads();
    const unsigned sel = selOr;

    unsigned rmask[8];
#pragma unroll
    for (int r = 0; r < 8; ++r)
        rmask[r] = mask[(size_t)bh * SEQ + qbase + r + hi8];

    FragB qa[2];                                     // Q 16x64 = 2 A-frags
    {
        int row = qbase + (lane & 15);
#pragma unroll
        for (int j = 0; j < 2; ++j)
#pragma unroll
            for (int v = 0; v < 8; ++v)
                qa[j].d[v] = *(const uint32_t*)&Qh[qvbase + (size_t)row * HDIM +
                                                  j * 32 + a_koff(v, lane)];
    }

    float m[8], l[8];
    v8f oacc[4] = {};
#pragma unroll
    for (int r = 0; r < 8; ++r) { m[r] = -3.0e38f; l[r] = 0.f; }

    for (int kb = 0; kb < NB; ++kb) {                // all blocks (mirror ref)
        bool active = (kb == cur) || ((sel >> kb) & 1u);
        if (!active) continue;                       // uniform across WG
        for (int hb = 0; hb < 2; ++hb) {
            const int key0 = kb * BLK + hb * 128;
#if defined(HAVE_TDM)
            if (wave == 0) {                         // TDM: one D# per tile
                unsigned ldsK = (unsigned)(size_t)
                    (__attribute__((address_space(3))) uint16_t*)Ks;
                unsigned ldsV = (unsigned)(size_t)
                    (__attribute__((address_space(3))) uint16_t*)Vs;
                // K^T tile: 64 rows (d) x 128 keys, row stride SEQ
                tdm_load_2d(ldsK, &Kt[ktbase + key0], 128, HDIM, SEQ, SEQ, HDIM);
                // V tile: 128 rows (key) x 64 d, row stride HDIM (contiguous)
                tdm_load_2d(ldsV, &Vh[qvbase + (size_t)key0 * HDIM],
                            HDIM, 128, HDIM, HDIM, SEQ);
                WAIT_TENSOR0();
            }
#elif defined(HAVE_ASYNC)
            {
                const uint16_t* kp = &Kt[ktbase + key0];
                const uint16_t* vp = &Vh[qvbase + (size_t)key0 * HDIM];
#pragma unroll
                for (int i = 0; i < 8; ++i) {
                    int n = threadIdx.x + i * 128;        // uint4 index
                    int krow = n >> 4, kc8 = (n & 15) * 8;
                    async_b128(&kp[(size_t)krow * SEQ + kc8], &Ks[krow * 128 + kc8]);
                    async_b128(&vp[n * 8], &Vs[n * 8]);
                }
                WAIT_ASYNC0();
            }
#else
            {
                const uint16_t* kp = &Kt[ktbase + key0];
                const uint16_t* vp = &Vh[qvbase + (size_t)key0 * HDIM];
#pragma unroll
                for (int i = 0; i < 8; ++i) {
                    int n = threadIdx.x + i * 128;
                    int krow = n >> 4, kc8 = (n & 15) * 8;
                    *(uint4*)&Ks[krow * 128 + kc8] =
                        *(const uint4*)&kp[(size_t)krow * SEQ + kc8];
                    *(uint4*)&Vs[n * 8] = *(const uint4*)&vp[n * 8];
                }
            }
#endif
            __syncthreads();
            for (int it = 0; it < 4; ++it) {         // 32 keys per iteration
                const int kabs0 = key0 + it * 32;
                v8f sfr[2];
#pragma unroll
                for (int half = 0; half < 2; ++half) {
                    v8f sacc = {};
                    int kcol = it * 32 + half * 16;  // key within staged 128
#pragma unroll
                    for (int ksx = 0; ksx < 2; ++ksx) {
                        FragB bf;                    // B = K^T: n = key (lane%16), k = d per element
                        int dabs = ksx * 32;         // d base of this k-step
#pragma unroll
                        for (int s = 0; s < 16; ++s)
                            bf.h[s] = Ks[(dabs + a_koff(s >> 1, lane) + (s & 1)) * 128 + kcol + (lane & 15)];
                        sacc = wmma_bf16(qa[ksx], bf, sacc);
                    }
                    sfr[half] = sacc;
                }
                // mask + online softmax (per C-layout row r)
#pragma unroll
                for (int r = 0; r < 8; ++r) {
                    int qrow = qbase + r + hi8;
#pragma unroll
                    for (int half = 0; half < 2; ++half) {
                        int key = kabs0 + half * 16 + (lane & 15);
                        bool ok = (kb == cur) ? (key <= qrow)
                                              : (((rmask[r] >> kb) & 1u) != 0);
                        sfr[half][r] = ok ? sfr[half][r] * 0.125f : -3.0e38f;
                    }
                    float t = fmaxf(sfr[0][r], sfr[1][r]);
#pragma unroll
                    for (int xm = 1; xm <= 8; xm <<= 1)
                        t = fmaxf(t, __shfl_xor(t, xm, 32));
                    float mn  = fmaxf(m[r], t);
                    float scl = __expf(m[r] - mn);
                    m[r] = mn;
                    float p0 = __expf(sfr[0][r] - mn);
                    float p1 = __expf(sfr[1][r] - mn);
                    float rs = p0 + p1;
#pragma unroll
                    for (int xm = 1; xm <= 8; xm <<= 1)
                        rs += __shfl_xor(rs, xm, 32);
                    l[r] = l[r] * scl + rs;
#pragma unroll
                    for (int nt = 0; nt < 4; ++nt) oacc[nt][r] *= scl;
                    int prow = r + hi8;
                    Ps[wave][prow * 32 + (lane & 15)]      = f2bf(p0);
                    Ps[wave][prow * 32 + 16 + (lane & 15)] = f2bf(p1);
                }
                FragB pa;                            // P 16x32 as A-frag
                {
                    int row = lane & 15;
#pragma unroll
                    for (int v = 0; v < 8; ++v)
                        pa.d[v] = *(const uint32_t*)&Ps[wave][row * 32 + a_koff(v, lane)];
                }
#pragma unroll
                for (int nt = 0; nt < 4; ++nt) {     // out += P * V
                    FragB vf;                        // B = V: n = d (lane%16), k = key per element
#pragma unroll
                    for (int s = 0; s < 16; ++s)
                        vf.h[s] = Vs[(it * 32 + a_koff(s >> 1, lane) + (s & 1)) * HDIM +
                                     nt * 16 + (lane & 15)];
                    oacc[nt] = wmma_bf16(pa, vf, oacc[nt]);
                }
            }
            __syncthreads();
        }
    }
    // epilogue: normalize + write [B,S,H*D] bf16 for the output GEMM
#pragma unroll
    for (int r = 0; r < 8; ++r) {
        float inv = 1.0f / l[r];
        int srow = qbase + r + hi8;
        size_t ob = ((size_t)(b * SEQ + srow)) * HIDDEN + h * HDIM;
#pragma unroll
        for (int nt = 0; nt < 4; ++nt)
            AO[ob + nt * 16 + (lane & 15)] = f2bf(oacc[nt][r] * inv);
    }
}

// ---------------------------------------------------------------------------
extern "C" void kernel_launch(void* const* d_in, const int* in_sizes, int n_in,
                              void* d_out, int out_size, void* d_ws, size_t ws_size,
                              hipStream_t stream) {
    const float* X  = (const float*)d_in[0];
    const float* Wq = (const float*)d_in[1];
    const float* Wk = (const float*)d_in[2];
    const float* Wv = (const float*)d_in[3];
    const float* Wo = (const float*)d_in[4];
    float* out = (float*)d_out;

    char* ws = (char*)d_ws;
    size_t off = 0;
    auto carve = [&](size_t bytes) -> void* {
        off = (off + 255) & ~(size_t)255;
        void* p = ws + off;
        off += bytes;
        return p;
    };
    uint16_t* Xb   = (uint16_t*)carve((size_t)ROWS * HIDDEN * 2);
    uint16_t* WqT  = (uint16_t*)carve((size_t)HIDDEN * HIDDEN * 2);
    uint16_t* WkT  = (uint16_t*)carve((size_t)HIDDEN * HIDDEN * 2);
    uint16_t* WvT  = (uint16_t*)carve((size_t)HIDDEN * HIDDEN * 2);
    uint16_t* WoT  = (uint16_t*)carve((size_t)HIDDEN * HIDDEN * 2);
    float*    tmpF = (float*)   carve((size_t)ROWS * HIDDEN * 4);
    uint16_t* Qhp  = (uint16_t*)carve((size_t)ROWS * HIDDEN * 2);
    uint16_t* Ktp  = (uint16_t*)carve((size_t)ROWS * HIDDEN * 2);  // transposed
    uint16_t* Vhp  = (uint16_t*)carve((size_t)ROWS * HIDDEN * 2);
    uint16_t* AOb  = (uint16_t*)carve((size_t)ROWS * HIDDEN * 2);
    float*    reps = (float*)   carve((size_t)BATCH * HEADS * NB * HDIM * 4);
    uint8_t*  mk   = (uint8_t*) carve((size_t)BATCH * HEADS * SEQ);

    const int NELT = ROWS * HIDDEN;                    // 3,145,728

    cvt_f32_bf16<<<NELT / 256, 256, 0, stream>>>(X, Xb, NELT);
    transposeW<<<(HIDDEN * HIDDEN) / 256, 256, 0, stream>>>(Wq, WqT);
    transposeW<<<(HIDDEN * HIDDEN) / 256, 256, 0, stream>>>(Wk, WkT);
    transposeW<<<(HIDDEN * HIDDEN) / 256, 256, 0, stream>>>(Wv, WvT);
    transposeW<<<(HIDDEN * HIDDEN) / 256, 256, 0, stream>>>(Wo, WoT);

    dim3 gg(HIDDEN / 64, ROWS / 128);                  // (12, 32)
    gemm_bf16<<<gg, 256, 0, stream>>>(Xb, WqT, tmpF, ROWS, HIDDEN, HIDDEN);
    rope_layout<<<NELT / 256, 256, 0, stream>>>(tmpF, Qhp, 1, 0);
    gemm_bf16<<<gg, 256, 0, stream>>>(Xb, WkT, tmpF, ROWS, HIDDEN, HIDDEN);
    rope_layout<<<NELT / 256, 256, 0, stream>>>(tmpF, Ktp, 1, 1);  // K transposed
    gemm_bf16<<<gg, 256, 0, stream>>>(Xb, WvT, tmpF, ROWS, HIDDEN, HIDDEN);
    rope_layout<<<NELT / 256, 256, 0, stream>>>(tmpF, Vhp, 0, 0);

    block_reps<<<(BATCH * HEADS * NB * HDIM) / 256, 256, 0, stream>>>(Ktp, reps);
    block_select<<<(BATCH * HEADS * SEQ) / 256, 256, 0, stream>>>(Qhp, reps, mk);

    dim3 ag(SEQ / 64, BATCH * HEADS);                  // (32, 24)
    moba_attn<<<ag, 128, 0, stream>>>(Qhp, Ktp, Vhp, mk, AOb);

    gemm_bf16<<<gg, 256, 0, stream>>>(AOb, WoT, out, ROWS, HIDDEN, HIDDEN);
}
